// StructureDecoder_26860725469613
// MI455X (gfx1250) — matrix operand order repaired and checked by reading, not compiled, not run on hardware
//
#include <hip/hip_runtime.h>
#include <hip/hip_bf16.h>
#include <math.h>

// ---- problem constants (match reference) ----
#define NN   8192      // nodes
#define DD   128       // feature dim
#define EE   262144    // edges

typedef _Float16 v16h __attribute__((ext_vector_type(16)));
typedef _Float16 v8h  __attribute__((ext_vector_type(8)));
typedef _Float16 v4h  __attribute__((ext_vector_type(4)));
typedef float    v8f  __attribute__((ext_vector_type(8)));

// ---------------- degree / normalization ----------------
__global__ void k_init_deg(float* __restrict__ deg) {
    int i = blockIdx.x * blockDim.x + threadIdx.x;
    if (i < NN) deg[i] = 1.0f;   // self-loop weight folded in
}

__global__ void k_deg_edges(const int* __restrict__ ei,
                            const float* __restrict__ ew,
                            float* __restrict__ deg) {
    int e = blockIdx.x * blockDim.x + threadIdx.x;
    if (e < EE) {
        int d = ei[EE + e];   // dst row of [2,E] edge_index (int32; JAX x64 off)
        __hip_atomic_fetch_add(&deg[d], ew[e], __ATOMIC_RELAXED, __HIP_MEMORY_SCOPE_AGENT);
    }
}

__global__ void k_dis(const float* __restrict__ deg, float* __restrict__ dis) {
    int i = blockIdx.x * blockDim.x + threadIdx.x;
    if (i < NN) {
        float d = deg[i];
        dis[i] = d > 0.0f ? rsqrtf(d) : 0.0f;
    }
}

// ---------------- h = x @ W  (tiny GEMM, one row per block) ----------------
__global__ void k_linear(const float* __restrict__ x,
                         const float* __restrict__ W,
                         float* __restrict__ h) {
    __shared__ float sx[DD];
    const int row = blockIdx.x;
    const int j   = threadIdx.x;
    sx[j] = x[(size_t)row * DD + j];
    __syncthreads();
    float acc = 0.0f;
#pragma unroll 8
    for (int k = 0; k < DD; ++k)
        acc = fmaf(sx[k], W[(size_t)k * DD + j], acc);
    h[(size_t)row * DD + j] = acc;
}

// ---------------- agg init: bias + self-loop message (float4) ----------------
__global__ void k_agg_init(const float* __restrict__ h,
                           const float* __restrict__ dis,
                           const float* __restrict__ b,
                           float* __restrict__ agg) {
    int t  = blockIdx.x * blockDim.x + threadIdx.x;   // over NN*DD/4
    int i4 = t * 4;
    int n  = i4 >> 7;
    float dn = dis[n];
    float s  = dn * dn;
    float4 hv = *(const float4*)&h[i4];
    float4 bv = *(const float4*)&b[i4 & (DD - 1)];
    float4 o;
    o.x = bv.x + s * hv.x;  o.y = bv.y + s * hv.y;
    o.z = bv.z + s * hv.z;  o.w = bv.w + s * hv.w;
    *(float4*)&agg[i4] = o;
}

// ---------------- edge scatter: one wave per edge, float4 + fp32 atomics ----
__global__ void k_scatter(const int* __restrict__ ei,
                          const float* __restrict__ ew,
                          const float* __restrict__ dis,
                          const float* __restrict__ h,
                          float* __restrict__ agg) {
    const int e    = (blockIdx.x * blockDim.x + threadIdx.x) >> 5; // global wave id
    const int lane = threadIdx.x & 31;
    const int s = ei[e];
    const int d = ei[EE + e];
    const float norm = dis[s] * ew[e] * dis[d];
    const float4 hv = *(const float4*)&h[(size_t)s * DD + lane * 4];
    float* dst = &agg[(size_t)d * DD + lane * 4];
    __hip_atomic_fetch_add(dst + 0, norm * hv.x, __ATOMIC_RELAXED, __HIP_MEMORY_SCOPE_AGENT);
    __hip_atomic_fetch_add(dst + 1, norm * hv.y, __ATOMIC_RELAXED, __HIP_MEMORY_SCOPE_AGENT);
    __hip_atomic_fetch_add(dst + 2, norm * hv.z, __ATOMIC_RELAXED, __HIP_MEMORY_SCOPE_AGENT);
    __hip_atomic_fetch_add(dst + 3, norm * hv.w, __ATOMIC_RELAXED, __HIP_MEMORY_SCOPE_AGENT);
}

// ---------------- ReLU + fp16 convert (float4 -> half4) ----------------
__global__ void k_relu_cvt(const float* __restrict__ agg, _Float16* __restrict__ zh) {
    int t  = blockIdx.x * blockDim.x + threadIdx.x;   // over NN*DD/4
    int i4 = t * 4;
    float4 a = *(const float4*)&agg[i4];
    v4h o;
    o[0] = (_Float16)fmaxf(a.x, 0.0f);
    o[1] = (_Float16)fmaxf(a.y, 0.0f);
    o[2] = (_Float16)fmaxf(a.z, 0.0f);
    o[3] = (_Float16)fmaxf(a.w, 0.0f);
    *(v4h*)&zh[i4] = o;
}

// ---------------- Gram matrix: out = Z @ Z^T with WMMA f16 ----------------
// 128x128 tile per block; 8 waves; each wave: 32x64 (2x4 accumulators of 16x16).
// K staged in 32-wide slabs, double-buffered in LDS via async global->LDS copies.
#define SLAB_W 32               // K per slab (halves)
#define SLAB_P 40               // padded row (halves): 80B stride -> conflict-free
__launch_bounds__(256)
__global__ void k_gram_wmma(const _Float16* __restrict__ zh, float* __restrict__ out) {
    __shared__ _Float16 sA[2][128][SLAB_P];   // 2 x 10 KB
    __shared__ _Float16 sB[2][128][SLAB_P];   // 2 x 10 KB  (total 40 KB)

    const int i0   = blockIdx.y * 128;
    const int j0   = blockIdx.x * 128;
    const int tid  = threadIdx.x;
    const int lane = tid & 31;
    const int wid  = tid >> 5;
    const int hl   = lane & 15;   // low-half lane id
    const int hi   = lane >> 4;   // which half of the wave
    const int rbase = (wid >> 1) * 32;   // wave row base in tile
    const int cbase = (wid & 1) * 64;    // wave col base in tile

    // panel-copy assignment: each thread moves 32B per panel per slab
    const int crow   = tid >> 1;          // 0..127
    const int cchunk = (tid & 1) * 16;    // halves (0 or 16) -> 32B

    v8f acc[2][4];
#pragma unroll
    for (int r = 0; r < 2; ++r)
#pragma unroll
        for (int c = 0; c < 4; ++c)
#pragma unroll
            for (int q = 0; q < 8; ++q) acc[r][c][q] = 0.0f;

    // async issue of one 32-wide K slab into buffer `buf`
    auto issue = [&](int ks, int buf) {
        const _Float16* gA = zh + (size_t)(i0 + crow) * DD + ks * SLAB_W + cchunk;
        const _Float16* gB = zh + (size_t)(j0 + crow) * DD + ks * SLAB_W + cchunk;
        unsigned la = (unsigned)(size_t)&sA[buf][crow][cchunk];
        unsigned lb = (unsigned)(size_t)&sB[buf][crow][cchunk];
        // INST_OFFSET applies to both LDS and global sides (ISA 15.18.3)
        asm volatile(
            "global_load_async_to_lds_b128 %0, %2, off\n\t"
            "global_load_async_to_lds_b128 %0, %2, off offset:16\n\t"
            "global_load_async_to_lds_b128 %1, %3, off\n\t"
            "global_load_async_to_lds_b128 %1, %3, off offset:16"
            :: "v"(la), "v"(lb), "v"(gA), "v"(gB)
            : "memory");
    };

    issue(0, 0);

#pragma unroll
    for (int ks = 0; ks < 4; ++ks) {
        const int buf = ks & 1;
        // own slab-ks async copies complete, then workgroup-wide rendezvous
        asm volatile("s_wait_asynccnt 0" ::: "memory");
        __syncthreads();
        if (ks + 1 < 4) issue(ks + 1, (ks + 1) & 1);   // prefetch next slab

        // A fragments: 16x32 f16 layout (lane<16: K 0-7 & 16-23; lane>=16: K 8-15 & 24-31)
        v16h afrag[2];
#pragma unroll
        for (int r = 0; r < 2; ++r) {
            const int M = rbase + r * 16 + hl;
            union { v16h v; v8h p[2]; } u;
            u.p[0] = *(const v8h*)&sA[buf][M][hi * 8];
            u.p[1] = *(const v8h*)&sA[buf][M][16 + hi * 8];
            afrag[r] = u.v;
        }
        // B fragments: 32x16 layout (lane<16: K 0-7 & 16-23; lane>=16: K 8-15 & 24-31; N = lane%16)
        v16h bfrag[4];
#pragma unroll
        for (int c = 0; c < 4; ++c) {
            const int Nr = cbase + c * 16 + hl;   // B[k][n] = Z[j0+Nr][k]
            union { v16h v; v8h p[2]; } u;
            u.p[0] = *(const v8h*)&sB[buf][Nr][hi * 8];
            u.p[1] = *(const v8h*)&sB[buf][Nr][16 + hi * 8];
            bfrag[c] = u.v;
        }
#pragma unroll
        for (int r = 0; r < 2; ++r)
#pragma unroll
            for (int c = 0; c < 4; ++c)
                acc[r][c] = __builtin_amdgcn_wmma_f32_16x16x32_f16(
                    false, afrag[r], false, bfrag[c],
                    (short)0, acc[r][c], false, false);
    }

    // epilogue: C/D layout -> VGPR v: rows v (lanes 0-15) / v+8 (lanes 16-31), col = lane%16
    // output is written once, never re-read: non-temporal stores keep L2 for zh
#pragma unroll
    for (int r = 0; r < 2; ++r)
#pragma unroll
        for (int c = 0; c < 4; ++c)
#pragma unroll
            for (int v = 0; v < 8; ++v) {
                const int row = i0 + rbase + r * 16 + v + hi * 8;
                const int col = j0 + cbase + c * 16 + hl;
                __builtin_nontemporal_store(acc[r][c][v], &out[(size_t)row * NN + col]);
            }
}

// ---------------- launch ----------------
extern "C" void kernel_launch(void* const* d_in, const int* in_sizes, int n_in,
                              void* d_out, int out_size, void* d_ws, size_t ws_size,
                              hipStream_t stream) {
    const float* x  = (const float*)d_in[0];
    const int*   ei = (const int*)d_in[1];    // [2,E] int32
    const float* ew = (const float*)d_in[2];
    const float* W  = (const float*)d_in[3];
    const float* b  = (const float*)d_in[4];
    float* out = (float*)d_out;

    char* ws = (char*)d_ws;
    float*    deg = (float*)(ws);                                  //  32 KB
    float*    dis = (float*)(ws + 32768);                          //  32 KB
    float*    h   = (float*)(ws + 65536);                          //   4 MB
    float*    agg = (float*)(ws + 65536 + 4194304);                //   4 MB
    _Float16* zh  = (_Float16*)(ws + 65536 + 2 * 4194304);         //   2 MB

    k_init_deg <<<NN / 256, 256, 0, stream>>>(deg);
    k_deg_edges<<<EE / 256, 256, 0, stream>>>(ei, ew, deg);
    k_dis      <<<NN / 256, 256, 0, stream>>>(deg, dis);
    k_linear   <<<NN, DD, 0, stream>>>(x, W, h);
    k_agg_init <<<(NN * DD / 4) / 256, 256, 0, stream>>>(h, dis, b, agg);
    k_scatter  <<<(EE * 32) / 256, 256, 0, stream>>>(ei, ew, dis, h, agg);
    k_relu_cvt <<<(NN * DD / 4) / 256, 256, 0, stream>>>(agg, zh);

    dim3 grid(NN / 128, NN / 128);
    k_gram_wmma<<<grid, 256, 0, stream>>>(zh, out);
}
